// LNN_54640573940240
// MI455X (gfx1250) — matrix operand order repaired and checked by reading, not compiled, not run on hardware
//
#include <hip/hip_runtime.h>
#include <hip/hip_bf16.h>

#define HID 512
#define SMP 8            // samples per workgroup
#define TPB 256          // 8 waves (wave32)

typedef __bf16 bf16_t;
typedef bf16_t v16bf __attribute__((ext_vector_type(16)));
typedef bf16_t v8bf  __attribute__((ext_vector_type(8)));
typedef float  v8f   __attribute__((ext_vector_type(8)));
typedef float  v4f   __attribute__((ext_vector_type(4)));

// A-operand slot -> K mapping (16-bit A matrix 16x32, ISA 7.12.2):
// slot e<8  -> k = 8*hf + e        (run 0, contiguous)
// slot e>=8 -> k = 16 + 8*hf + e-8 (run 1, contiguous)
__device__ __forceinline__ int kidxA(int e, int hf) {
  return (e < 8) ? (8 * hf + e) : (8 + 8 * hf + e);
}

__device__ __forceinline__ v8f wmma_bf16(v16bf a, v16bf b, v8f c) {
  return __builtin_amdgcn_wmma_f32_16x16x32_bf16(false, a, false, b, (short)0, c, false, false);
}

__device__ __forceinline__ v16bf pack16(v4f p0, v4f p1, v4f p2, v4f p3) {
  v16bf r;
#pragma unroll
  for (int e = 0; e < 4; ++e) {
    r[e]      = (bf16_t)p0[e];
    r[4 + e]  = (bf16_t)p1[e];
    r[8 + e]  = (bf16_t)p2[e];
    r[12 + e] = (bf16_t)p3[e];
  }
  return r;
}

// A operand from a bf16 row: two 16B LDS loads, no conversion.
__device__ __forceinline__ v16bf loadA16(const bf16_t* row, int c0) {
  v8bf lo = *(const v8bf*)(row + c0);
  v8bf hi = *(const v8bf*)(row + c0 + 16);
  return __builtin_shufflevector(lo, hi, 0, 1, 2, 3, 4, 5, 6, 7,
                                 8, 9, 10, 11, 12, 13, 14, 15);
}

// A operand = packed f32 weights (global, A-slot order) * f32 scale row (LDS).
__device__ __forceinline__ v16bf mulA16(const float* wa, const float* srow, int c0) {
  v4f w0 = *(const v4f*)(wa);
  v4f w1 = *(const v4f*)(wa + 4);
  v4f w2 = *(const v4f*)(wa + 8);
  v4f w3 = *(const v4f*)(wa + 12);
  v4f s0 = *(const v4f*)(srow + c0);
  v4f s1 = *(const v4f*)(srow + c0 + 4);
  v4f s2 = *(const v4f*)(srow + c0 + 16);
  v4f s3 = *(const v4f*)(srow + c0 + 20);
  return pack16(w0 * s0, w1 * s1, w2 * s2, w3 * s3);
}

__device__ __forceinline__ float sigmoidf_(float z) { return 1.f / (1.f + __expf(-z)); }
__device__ __forceinline__ float softplusf_(float z) { return (z > 20.f) ? z : log1pf(__expf(z)); }

// ---------------------------------------------------------------------------
// Prep: pack W2, W2^T, W1q^T into WMMA B-operand tile order (bf16) and W1q
// into A-operand slot order (f32) so hot loops do coalesced b128 loads.
// ---------------------------------------------------------------------------
__global__ __launch_bounds__(TPB) void lnn_prep(
    const float* __restrict__ W1, const float* __restrict__ W2,
    bf16_t* __restrict__ W2pk, bf16_t* __restrict__ W2tpk,
    bf16_t* __restrict__ W1qtpk, float* __restrict__ W1apk)
{
  int tid = blockIdx.x * blockDim.x + threadIdx.x;
  int nth = gridDim.x * blockDim.x;
  for (int idx = tid; idx < 16 * 32 * 32 * 16; idx += nth) {
    int e  = idx & 15;
    int ln = (idx >> 4) & 31;
    int nt = (idx >> 9) & 31;
    int kt = idx >> 14;
    int kg = 32 * kt + kidxA(e, ln >> 4);
    int ng = 16 * nt + (ln & 15);
    W2pk[idx]  = (bf16_t)W2[kg * HID + ng];   // B = W2
    W2tpk[idx] = (bf16_t)W2[ng * HID + kg];   // B = W2^T
  }
  for (int idx = tid; idx < 16 * 32 * 16; idx += nth) {
    int e  = idx & 15;
    int ln = (idx >> 4) & 31;
    int kt = idx >> 9;
    int kg = 32 * kt + kidxA(e, ln >> 4);
    W1qtpk[idx] = (bf16_t)W1[(16 + (ln & 15)) * HID + kg];   // B = W1q^T
    int ka = 32 * kt + kidxA(e, ln >> 4);
    W1apk[idx]  = W1[(16 + (ln & 15)) * HID + ka];           // A-layout W1q (f32)
  }
}

// ---------------------------------------------------------------------------
// Main: one workgroup = 8 samples = 8 waves; wave w owns sample w.
// ---------------------------------------------------------------------------
__global__ __launch_bounds__(TPB) void lnn_main(
    const float* __restrict__ x,  const float* __restrict__ W1,
    const float* __restrict__ b1, const float* __restrict__ b2,
    const float* __restrict__ w3,
    const bf16_t* __restrict__ W2pk, const bf16_t* __restrict__ W2tpk,
    const bf16_t* __restrict__ W1qtpk, const float* __restrict__ W1apk,
    float* __restrict__ out)
{
  __shared__ __align__(16) float s1m [SMP][HID];   // sigmoid(z1)
  __shared__ __align__(16) float d2f [SMP][HID];   // d2 = w3*s2*(1-s2)
  __shared__ __align__(16) float zvd [SMP][HID];   // z2 -> v -> d1
  __shared__ __align__(16) float pbuf[SMP][HID];   // p_i = sum_k qd_k W1[k,i]
  __shared__ __align__(16) bf16_t abU2[16][HID];   // u2 (rows 8..15 zero)
  __shared__ __align__(16) bf16_t abC [16][HID];   // c = p.*s1 (rows 8..15 zero)
  // unions: abA1 (a1, phase 1) then rbuf (phases 5..6); tileb (ph 6) then solver (ph 8)
  __shared__ __align__(16) unsigned char uAbuf[16 * HID * sizeof(bf16_t)];
  __shared__ __align__(16) float uS[SMP][288];
  __shared__ float xsh[SMP][32];
  __shared__ float tvec[SMP][16];
  __shared__ float dlq [SMP][16];

  bf16_t (*abA1)[HID] = reinterpret_cast<bf16_t (*)[HID]>(uAbuf);
  float  (*rbuf)[HID] = reinterpret_cast<float  (*)[HID]>(uAbuf);

  const int tid  = threadIdx.x;
  const int w    = tid >> 5;
  const int ln   = tid & 31;
  const int hf   = ln >> 4;
  const int lm   = ln & 15;
  const int c0b  = 8 * hf;
  const int base = blockIdx.x * SMP;

  // ---- Phase 0: z1, s1, a1, p, c -----------------------------------------
  xsh[w][ln] = x[(base + w) * 32 + ln];
  for (int i = ln; i < HID; i += 32) {     // zero-pad rows 8..15 of A staging
    abA1[8 + w][i] = (bf16_t)0.f;
    abU2[8 + w][i] = (bf16_t)0.f;
    abC [8 + w][i] = (bf16_t)0.f;
  }
  __syncthreads();
  for (int t = 0; t < 16; ++t) {
    int i = ln + 32 * t;
    float z = b1[i];
    float p = 0.f;
#pragma unroll
    for (int k = 0; k < 16; ++k) {
      float wq  = W1[k * HID + i];
      float wqd = W1[(16 + k) * HID + i];
      z += xsh[w][k] * wq + xsh[w][16 + k] * wqd;
      p += xsh[w][16 + k] * wq;
    }
    float sg = sigmoidf_(z);
    s1m[w][i]  = sg;
    pbuf[w][i] = p;
    abA1[w][i] = (bf16_t)softplusf_(z);
    abC [w][i] = (bf16_t)(p * sg);
  }
  __syncthreads();

  // ---- Phase 1: z2 = a1 @ W2 + b2 ----------------------------------------
  {
    v16bf areg[16];
#pragma unroll
    for (int kt = 0; kt < 16; ++kt) areg[kt] = loadA16(abA1[lm], 32 * kt + c0b);
    for (int nti = 0; nti < 4; ++nti) {
      int nt = w + 8 * nti;
      v8f acc = {};
#pragma unroll
      for (int kt = 0; kt < 16; ++kt) {
        v16bf bv = *(const v16bf*)(W2pk + ((kt * 32 + nt) * 32 + ln) * 16);
        acc = wmma_bf16(areg[kt], bv, acc);
      }
      if (hf == 0) {
        int n = 16 * nt + lm;
        float bias = b2[n];
#pragma unroll
        for (int t = 0; t < 8; ++t) zvd[t][n] = acc[t] + bias;
      }
    }
  }
  __syncthreads();

  // ---- Phase 2: u2 = s2.*w3 (bf16), d2 = w3.*s2.*(1-s2) ------------------
  for (int t = 0; t < 16; ++t) {
    int i = ln + 32 * t;
    float s2 = sigmoidf_(zvd[w][i]);
    float w3v = w3[i];
    abU2[w][i] = (bf16_t)(s2 * w3v);
    d2f[w][i]  = w3v * s2 * (1.f - s2);
  }
  __syncthreads();

  // ---- Phase 3: v = u2 @ W2^T --------------------------------------------
  {
    v16bf areg[16];
#pragma unroll
    for (int kt = 0; kt < 16; ++kt) areg[kt] = loadA16(abU2[lm], 32 * kt + c0b);
    for (int nti = 0; nti < 4; ++nti) {
      int nt = w + 8 * nti;
      v8f acc = {};
#pragma unroll
      for (int kt = 0; kt < 16; ++kt) {
        v16bf bv = *(const v16bf*)(W2tpk + ((kt * 32 + nt) * 32 + ln) * 16);
        acc = wmma_bf16(areg[kt], bv, acc);
      }
      if (hf == 0) {
        int n = 16 * nt + lm;
#pragma unroll
        for (int t = 0; t < 8; ++t) zvd[t][n] = acc[t];
      }
    }
  }
  __syncthreads();

  // ---- Phase 4: DL_q, T1 (vectorized), then d1 in place ------------------
  if (ln < 16) {
    const float* w1q  = W1 + ln * HID;
    const float* w1qd = W1 + (16 + ln) * HID;
    v4f adl4 = {}, at14 = {};
    for (int i = 0; i < HID; i += 4) {
      v4f sg = *(const v4f*)&s1m[w][i];
      v4f vv = *(const v4f*)&zvd[w][i];
      v4f pp = *(const v4f*)&pbuf[w][i];
      v4f wa = *(const v4f*)(w1q + i);
      v4f wb = *(const v4f*)(w1qd + i);
      v4f sv = sg * vv;
      adl4 += wa * sv;
      at14 += pp * sv * (1.f - sg) * wb;
    }
    dlq[w][ln]  = adl4[0] + adl4[1] + adl4[2] + adl4[3];
    tvec[w][ln] = at14[0] + at14[1] + at14[2] + at14[3];
  }
  for (int i0 = 4 * ln; i0 < HID; i0 += 128) {   // d1 = s1*(1-s1)*v
    v4f sg = *(const v4f*)&s1m[w][i0];
    v4f vv = *(const v4f*)&zvd[w][i0];
    *(v4f*)&zvd[w][i0] = sg * (1.f - sg) * vv;
  }
  __syncthreads();

  // ---- Phase 5: r = (p .* s1) @ W2 ---------------------------------------
  {
    v16bf areg[16];
#pragma unroll
    for (int kt = 0; kt < 16; ++kt) areg[kt] = loadA16(abC[lm], 32 * kt + c0b);
    for (int nti = 0; nti < 4; ++nti) {
      int nt = w + 8 * nti;
      v8f acc = {};
#pragma unroll
      for (int kt = 0; kt < 16; ++kt) {
        v16bf bv = *(const v16bf*)(W2pk + ((kt * 32 + nt) * 32 + ln) * 16);
        acc = wmma_bf16(areg[kt], bv, acc);
      }
      if (hf == 0) {
        int n = 16 * nt + lm;
#pragma unroll
        for (int t = 0; t < 8; ++t) rbuf[t][n] = acc[t];
      }
    }
  }
  __syncthreads();

  // ---- Phase 6: Aq = (W1q.*s1) @ W2 fused into hqq and T2 ----------------
  float* tb = uS[w];
  v8f hqq = {};
  float t2acc = 0.f;
  {
    v16bf aq[16];                      // A fragments, invariant across nt
#pragma unroll
    for (int kt = 0; kt < 16; ++kt)
      aq[kt] = mulA16(W1apk + (kt * 32 + ln) * 16, s1m[w], 32 * kt + c0b);

    for (int nt = 0; nt < 32; ++nt) {
      v8f acc = {};
#pragma unroll
      for (int kt = 0; kt < 16; ++kt) {
        v16bf bv = *(const v16bf*)(W2pk + ((kt * 32 + nt) * 32 + ln) * 16);
        acc = wmma_bf16(aq[kt], bv, acc);
      }
#pragma unroll
      for (int t = 0; t < 8; ++t) tb[(8 * hf + t) * 16 + lm] = acc[t];
      __syncthreads();
      int i0 = 16 * nt;
      // A2 = (Aq .* d2), valid K slots are e<8 (k = 8*hf+e < 16)
      v4f t0  = *(const v4f*)&tb[lm * 16 + 8 * hf];
      v4f t1  = *(const v4f*)&tb[lm * 16 + 8 * hf + 4];
      v4f dd0 = *(const v4f*)&d2f[w][i0 + 8 * hf];
      v4f dd1 = *(const v4f*)&d2f[w][i0 + 8 * hf + 4];
      v4f q0 = t0 * dd0, q1 = t1 * dd1;
      v16bf a2;
#pragma unroll
      for (int e = 0; e < 4; ++e) {
        a2[e]      = (bf16_t)q0[e];
        a2[4 + e]  = (bf16_t)q1[e];
        a2[8 + e]  = (bf16_t)0.f;
        a2[12 + e] = (bf16_t)0.f;
      }
      // B2 = Aq^T in A-slot order: slots e<8 -> k = 8*hf+e, slots e>=8 (k>=16) zero
      float msk = (hf == 0) ? 1.f : 0.f;
      v4f r0 = *(const v4f*)&tb[lm * 16 + 8 * hf];
      v4f r1 = *(const v4f*)&tb[lm * 16 + 8 * hf + 4];
      v4f r2 = *(const v4f*)&tb[lm * 16 + 8]  * 0.f;
      v4f r3 = *(const v4f*)&tb[lm * 16 + 12] * 0.f;
      hqq = wmma_bf16(a2, pack16(r0, r1, r2, r3), hqq);
      if (hf == 0) {
        v4f s4 = {};
#pragma unroll
        for (int c = 0; c < 4; ++c) {
          v4f rr = *(const v4f*)&rbuf[w][i0 + 4 * c];
          v4f dd = *(const v4f*)&d2f[w][i0 + 4 * c];
          v4f tt = *(const v4f*)&tb[lm * 16 + 4 * c];
          s4 += rr * dd * tt;
        }
        t2acc += s4[0] + s4[1] + s4[2] + s4[3];
      }
      __syncthreads();
    }
  }

  // ---- Phase 7: hqq += (W1q .* d1) @ W1q^T -------------------------------
  for (int kt = 0; kt < 16; ++kt) {
    v16bf av = mulA16(W1apk + (kt * 32 + ln) * 16, zvd[w], 32 * kt + c0b);
    v16bf bv = *(const v16bf*)(W1qtpk + (kt * 32 + ln) * 16);
    hqq = wmma_bf16(av, bv, hqq);
  }

  // ---- Phase 8: scatter Hqq + rhs, pivoted Gauss-Jordan solve ------------
  float* sv = uS[w];          // reuse tile region as 16x(17) solver, stride 18
#pragma unroll
  for (int t = 0; t < 8; ++t) sv[(8 * hf + t) * 18 + lm] = hqq[t];
  if (hf == 0) tvec[w][lm] += t2acc;
  if (ln < 16) sv[ln * 18 + 16] = dlq[w][ln] - tvec[w][ln];
  __syncthreads();

  volatile float* M = sv;
  for (int k = 0; k < 16; ++k) {
    int piv = k;                              // redundant uniform pivot scan
    float best = fabsf(M[k * 18 + k]);
    for (int r = k + 1; r < 16; ++r) {
      float a = fabsf(M[r * 18 + k]);
      if (a > best) { best = a; piv = r; }
    }
    if (ln < 17 && piv != k) {
      float tmp = M[k * 18 + ln];
      M[k * 18 + ln] = M[piv * 18 + ln];
      M[piv * 18 + ln] = tmp;
    }
    __syncthreads();
    float inv = 1.f / M[k * 18 + k];
    float rowk = 0.f;
    if (ln < 17) { rowk = M[k * 18 + ln] * inv; M[k * 18 + ln] = rowk; }
    __syncthreads();
    if (ln < 17) {
      for (int r = 0; r < 16; ++r) {
        if (r == k) continue;
        float f = M[r * 18 + k];
        M[r * 18 + ln] -= f * rowk;
      }
    }
    __syncthreads();
  }

  // ---- Output: [qd, qdd] -------------------------------------------------
  if (ln < 16) {
    out[(base + w) * 32 + ln]      = xsh[w][16 + ln];
    out[(base + w) * 32 + 16 + ln] = M[ln * 18 + 16];
  }
}

// ---------------------------------------------------------------------------
extern "C" void kernel_launch(void* const* d_in, const int* in_sizes, int n_in,
                              void* d_out, int out_size, void* d_ws, size_t ws_size,
                              hipStream_t stream) {
  const float* x  = (const float*)d_in[0];
  const float* W1 = (const float*)d_in[1];
  const float* b1 = (const float*)d_in[2];
  const float* W2 = (const float*)d_in[3];
  const float* b2 = (const float*)d_in[4];
  const float* W3 = (const float*)d_in[5];
  // d_in[6] = b3: drops out of all derivatives.

  char* ws = (char*)d_ws;
  bf16_t* W2pk   = (bf16_t*)(ws);                     // 512 KB
  bf16_t* W2tpk  = (bf16_t*)(ws + (512u << 10));      // 512 KB
  bf16_t* W1qtpk = (bf16_t*)(ws + (1024u << 10));     // 16 KB
  float*  W1apk  = (float*) (ws + (1040u << 10));     // 32 KB

  lnn_prep<<<256, TPB, 0, stream>>>(W1, W2, W2pk, W2tpk, W1qtpk, W1apk);

  int nsamples = out_size / 32;                       // B = 2048
  lnn_main<<<nsamples / SMP, TPB, 0, stream>>>(x, W1, b1, b2, W3,
                                               W2pk, W2tpk, W1qtpk, W1apk,
                                               (float*)d_out);
  (void)in_sizes; (void)n_in; (void)ws_size;
}
